// OriginalConjunctionLayer_63995012710755
// MI455X (gfx1250) — matrix operand order repaired and checked by reading, not compiled, not run on hardware
//
#include <hip/hip_runtime.h>
#include <hip/hip_bf16.h>
#include <stdint.h>
#include <float.h>

// ---- WMMA fragment types (CDNA5 gfx1250, wave32) ----
typedef __attribute__((ext_vector_type(16))) __bf16 v16bf;
typedef __attribute__((ext_vector_type(8)))  float  v8f;

#define D_DIM 1024
#define N_DIM 64
#define B_DIM 4096
#define KC_STEPS (D_DIM / 32)   // 32 K-chunks of 32

__device__ __forceinline__ __bf16 lg2bf(float v) {
    // clamp to [1e-30, FLT_MAX] in one v_med3_f32, then v_log_f32 (base-2)
    return (__bf16)__builtin_amdgcn_logf(__builtin_amdgcn_fmed3f(v, 1e-30f, FLT_MAX));
}

// -------------------------------------------------------------------------
// Kernel 1: binarize W (D x N f32 row-major) and write bf16 0/1 pairs in
// B-FRAGMENT ORDER so the GEMM kernel loads each fragment with two b128s.
// Layout (dwords): index = ((kc*4 + ntile)*32 + lane)*8 + j
//   lane = kgroup*16 + (n % 16), kgroup = lane>>4
//   dword j = { bf16(Wb[K+1, n]) , bf16(Wb[K, n]) },  K = kc*32 + kgroup*8 + (j<4 ? 2j : 16 + 2(j-4))
// bf16(1.0) == 0x3F80 -> pure integer packing.
// -------------------------------------------------------------------------
__global__ __launch_bounds__(256) void binarize_pack_W(
    const float* __restrict__ W, uint32_t* __restrict__ Wp)
{
    int i = blockIdx.x * blockDim.x + threadIdx.x;   // 0 .. 32767
    if (i >= KC_STEPS * 4 * 32 * 8) return;
    const int j      = i & 7;
    const int lane   = (i >> 3) & 31;
    const int nt     = (i >> 8) & 3;
    const int kc     = i >> 10;
    const int kgroup = lane >> 4;
    const int n      = nt * 16 + (lane & 15);
    const int k      = kc * 32 + kgroup * 8 + ((j < 4) ? 2 * j : 16 + 2 * (j - 4));
    uint32_t lo = (W[(size_t)k       * N_DIM + n] > 0.5f) ? 0x3F80u : 0u;
    uint32_t hi = (W[(size_t)(k + 1) * N_DIM + n] > 0.5f) ? 0x3F80u : 0u;
    Wp[i] = lo | (hi << 16);
}

// -------------------------------------------------------------------------
// Kernel 2: log2-space GEMM via v_wmma_f32_16x16x32_bf16.
//   res_bar = exp2( log2(clamp(x)) @ Wb ) ; m_out = 1
// One wave per 16-row tile, 4 accumulators cover N=64, so each x element
// pays one v_log_f32. A loaded as 4 x b128, each B fragment as 2 x b128.
// -------------------------------------------------------------------------
__global__ __launch_bounds__(256) void conj_wmma_kernel(
    const float* __restrict__ x,
    const uint32_t* __restrict__ Wp,
    float* __restrict__ res_out,
    float* __restrict__ m_out)
{
    const int lane     = threadIdx.x & 31;
    const int wave     = threadIdx.x >> 5;                 // 0..7
    const int row_tile = blockIdx.x * 8 + wave;            // 0..255
    const int m0       = row_tile * 16;
    const int lrow     = lane & 15;                        // 0..15
    const int kgroup   = lane >> 4;                        // 0 or 1

    const float* __restrict__ xrow = x + (size_t)(m0 + lrow) * D_DIM + kgroup * 8;

    v8f acc0 = {}, acc1 = {}, acc2 = {}, acc3 = {};

    for (int kc = 0; kc < KC_STEPS; ++kc) {
        const int k0 = kc * 32;

        // ---- A fragment (16x32 bf16): 4 contiguous float4 loads per lane ----
        const float4 a0 = *(const float4*)(xrow + k0);          // K = kg*8 + 0..3
        const float4 a1 = *(const float4*)(xrow + k0 + 4);      // K = kg*8 + 4..7
        const float4 a2 = *(const float4*)(xrow + k0 + 16);     // K = 16+kg*8+0..3
        const float4 a3 = *(const float4*)(xrow + k0 + 20);     // K = 16+kg*8+4..7

        union { v16bf v; __bf16 e[16]; } A;
        A.e[0]  = lg2bf(a0.x); A.e[1]  = lg2bf(a0.y);
        A.e[2]  = lg2bf(a0.z); A.e[3]  = lg2bf(a0.w);
        A.e[4]  = lg2bf(a1.x); A.e[5]  = lg2bf(a1.y);
        A.e[6]  = lg2bf(a1.z); A.e[7]  = lg2bf(a1.w);
        A.e[8]  = lg2bf(a2.x); A.e[9]  = lg2bf(a2.y);
        A.e[10] = lg2bf(a2.z); A.e[11] = lg2bf(a2.w);
        A.e[12] = lg2bf(a3.x); A.e[13] = lg2bf(a3.y);
        A.e[14] = lg2bf(a3.z); A.e[15] = lg2bf(a3.w);

        // ---- 4 B fragments: 8 contiguous dwords per lane each (2 x b128) ----
        const uint32_t* bp = Wp + (size_t)kc * 1024 + (size_t)lane * 8;
        union { v16bf v; uint4 q[2]; } B0, B1, B2, B3;
        B0.q[0] = *(const uint4*)(bp);         B0.q[1] = *(const uint4*)(bp + 4);
        B1.q[0] = *(const uint4*)(bp + 256);   B1.q[1] = *(const uint4*)(bp + 260);
        B2.q[0] = *(const uint4*)(bp + 512);   B2.q[1] = *(const uint4*)(bp + 516);
        B3.q[0] = *(const uint4*)(bp + 768);   B3.q[1] = *(const uint4*)(bp + 772);

        // prefetch next A stripe into cache (global_prefetch_b8)
        if (kc + 1 < KC_STEPS) __builtin_prefetch(xrow + k0 + 32, 0, 3);

        acc0 = __builtin_amdgcn_wmma_f32_16x16x32_bf16(false, A.v, false, B0.v,
                                                       (short)0, acc0, false, false);
        acc1 = __builtin_amdgcn_wmma_f32_16x16x32_bf16(false, A.v, false, B1.v,
                                                       (short)0, acc1, false, false);
        acc2 = __builtin_amdgcn_wmma_f32_16x16x32_bf16(false, A.v, false, B2.v,
                                                       (short)0, acc2, false, false);
        acc3 = __builtin_amdgcn_wmma_f32_16x16x32_bf16(false, A.v, false, B3.v,
                                                       (short)0, acc3, false, false);
    }

    // ---- epilogue: exp2 out of log space; C/D 16x16 f32 layout ----
    const int col     = lrow;
    const int rowbase = m0 + 8 * kgroup;
#pragma unroll
    for (int j = 0; j < 8; ++j) {
        const size_t row = (size_t)(rowbase + j);
        float* o = res_out + row * N_DIM + col;
        o[0]  = __builtin_amdgcn_exp2f(acc0[j]);
        o[16] = __builtin_amdgcn_exp2f(acc1[j]);
        o[32] = __builtin_amdgcn_exp2f(acc2[j]);
        o[48] = __builtin_amdgcn_exp2f(acc3[j]);
    }

    // ---- m_out = 1.0 : constant fill, fully vectorized b128 stores ----
    // Wave's tile = rows [m0, m0+16) x 64 cols = 1024 floats = 256 float4.
    const float4 one4 = make_float4(1.0f, 1.0f, 1.0f, 1.0f);
    float4* mo = (float4*)(m_out + (size_t)m0 * N_DIM);
#pragma unroll
    for (int j = 0; j < 8; ++j)
        mo[j * 32 + lane] = one4;
}

// -------------------------------------------------------------------------
extern "C" void kernel_launch(void* const* d_in, const int* in_sizes, int n_in,
                              void* d_out, int out_size, void* d_ws, size_t ws_size,
                              hipStream_t stream) {
    const float* x = (const float*)d_in[0];
    // d_in[1] = m : unused (missing_aware=False) -> never read (saves 16 MB traffic)
    const float* W = (const float*)d_in[2];

    float*    out = (float*)d_out;                 // [0, B*N): res_bar ; [B*N, 2*B*N): m_out
    uint32_t* Wp  = (uint32_t*)d_ws;               // 32768 dwords = 128 KB

    const int packElems = KC_STEPS * 4 * 32 * 8;   // 32768
    binarize_pack_W<<<(packElems + 255) / 256, 256, 0, stream>>>(W, Wp);

    // 4096/16 = 256 row tiles; 8 waves per 256-thread block -> 32 blocks
    conj_wmma_kernel<<<B_DIM / 16 / 8, 256, 0, stream>>>(
        x, Wp, out, out + (size_t)B_DIM * N_DIM);
}
